// SimCLRManifoldLoss_28312424415849
// MI455X (gfx1250) — matrix operand order repaired and checked by reading, not compiled, not run on hardware
//
#include <hip/hip_runtime.h>
#include <hip/hip_bf16.h>

#define NROW 8192          // N = B*V
#define BB   4096          // batch
#define DD   512           // feature dim
#define INVT 10.0f         // 1/temperature
#define KSTEPS 16          // 512 / 32 (K per bf16 WMMA)
#define NTILES 512         // 8192 / 16 column tiles total
#define SPLIT  4           // column splits (occupancy)
#define CTILES (NTILES / SPLIT)   // 128 column tiles per wave
#define LDS_STRIDE (DD + 8)       // 520 ushorts per staged column (matches TDM pad)

typedef __attribute__((ext_vector_type(16))) __bf16         v16bf;
typedef __attribute__((ext_vector_type(8)))  float          v8f;
typedef __attribute__((ext_vector_type(8)))  unsigned short v8u;
typedef __attribute__((ext_vector_type(4)))  unsigned int   v4ui;
typedef __attribute__((ext_vector_type(8)))  int            v8i;
typedef __attribute__((ext_vector_type(4)))  int            v4i;

union BFrag {
    v16bf b;
    struct { v8u lo; v8u hi; } h;
};

__device__ __forceinline__ unsigned short to_bf16_rne(float f) {
    unsigned int u = __float_as_uint(f);
    u += 0x7FFFu + ((u >> 16) & 1u);      // round-to-nearest-even
    return (unsigned short)(u >> 16);
}

// ---------------- Kernel 1: row-normalize + convert to bf16 ----------------
__global__ void k_normalize(const float* __restrict__ feat,
                            unsigned short* __restrict__ fb) {
    __shared__ float red[256];
    const int row = blockIdx.x;
    const int t   = threadIdx.x;
    const float* fr = feat + (size_t)row * DD;
    float a0 = fr[t];
    float a1 = fr[t + 256];
    red[t] = a0 * a0 + a1 * a1;
    __syncthreads();
    #pragma unroll
    for (int s = 128; s > 0; s >>= 1) {
        if (t < s) red[t] += red[t + s];
        __syncthreads();
    }
    const float inv = rsqrtf(red[0]);
    unsigned short* dst = fb + (size_t)row * DD;
    dst[t]       = to_bf16_rne(a0 * inv);
    dst[t + 256] = to_bf16_rne(a1 * inv);
}

// -------- TDM: DMA one 16-row x 512-col bf16 panel into LDS (padded) --------
// D# per CDNA5 ISA ch.8: group0 = {count/flags, lds_addr, global_addr, type},
// group1 = {data_size|pad cfg, tensor_dim0/1, tile_dim0/1/2, dim0_stride}.
// pad_interval=7 (every 256 DWORDs = one 512-bf16 row), pad_amount=3 (4 DWORDs)
// => hardware reproduces the 520-ushort LDS row stride for bank spreading.
__device__ __forceinline__ void tdm_load_tile(const unsigned short* src,
                                              unsigned int lds_off) {
    unsigned long long ga = (unsigned long long)(uintptr_t)src;
    v4ui g0;
    g0[0] = 1u;                                            // count=1 (valid), user mode
    g0[1] = lds_off;                                       // lds_addr (bytes)
    g0[2] = (unsigned int)(ga & 0xFFFFFFFFu);              // global_addr[31:0]
    g0[3] = (unsigned int)((ga >> 32) & 0x1FFFFFFu)        // global_addr[56:32]
          | (2u << 30);                                    // type=2 ("image")
    v8i g1;
    g1[0] = (int)((1u << 16)      // data_size = 2 bytes
                | (1u << 20)      // pad_enable
                | (7u << 22)      // pad_interval: 256 DWORDs
                | (3u << 25));    // pad_amount:   4 DWORDs
    g1[1] = (int)(512u << 16);    // [63:48]=tensor_dim0 lo = 512
    g1[2] = (int)(8192u << 16);   // [79:64]=dim0 hi=0, [95:80]=tensor_dim1 lo = 8192
    g1[3] = (int)(512u << 16);    // [111:96]=dim1 hi=0, [127:112]=tile_dim0 = 512
    g1[4] = 16;                   // [143:128]=tile_dim1 = 16, tile_dim2 = 0
    g1[5] = 512;                  // tensor_dim0_stride[31:0] = 512 elements
    g1[6] = 0;                    // stride hi / dim1_stride lo
    g1[7] = 0;
    v4i z4 = {0, 0, 0, 0};            // groups 2/3 unused (2D tensor)
    v8i z8 = {0, 0, 0, 0, 0, 0, 0, 0};
    __builtin_amdgcn_tensor_load_to_lds(g0, g1, z4, z4, z8, 0);
}

// -------- Kernel 2: fused bf16-WMMA Gram + streaming exp-sum per row --------
__global__ __launch_bounds__(256)
void k_gram_lse(const unsigned short* __restrict__ fb,
                float* __restrict__ rowsum_p,   // [SPLIT][NROW]
                float* __restrict__ posv_p,     // [SPLIT][NROW]
                float* __restrict__ diagv_p) {  // [SPLIT][NROW]
    __shared__ unsigned short bsh[2][16 * LDS_STRIDE]; // double-buffered B panel

    const int t     = threadIdx.x;
    const int wave  = t >> 5;
    const int lane  = t & 31;
    const int g     = lane >> 4;   // lane group (0/1)
    const int ln    = lane & 15;
    const int rg    = blockIdx.x & 63;        // row group: 8 row-tiles
    const int split = blockIdx.x >> 6;        // column split 0..3
    const int I0    = (rg * 8 + wave) * 16;   // this wave's 16 rows
    const int jt0   = split * CTILES;         // first column tile

    // ---- preload A panel: 16 rows x 512 K in bf16, contiguous K = 16g + s order ----
    v16bf afrag[KSTEPS];
    {
        const unsigned short* arow = fb + (size_t)(I0 + ln) * DD;
        #pragma unroll
        for (int kk = 0; kk < KSTEPS; ++kk) {
            const int kb = kk * 32;
            BFrag f;
            f.h.lo = *(const v8u*)(arow + kb + 16 * g);       // K = 16g + [0..7]
            f.h.hi = *(const v8u*)(arow + kb + 16 * g + 8);   // K = 16g + 8 + [0..7]
            afrag[kk] = f.b;
        }
    }

    float acc[8] = {0.f, 0.f, 0.f, 0.f, 0.f, 0.f, 0.f, 0.f};
    float pos[8] = {0.f, 0.f, 0.f, 0.f, 0.f, 0.f, 0.f, 0.f};
    float dia[8] = {0.f, 0.f, 0.f, 0.f, 0.f, 0.f, 0.f, 0.f};

    const unsigned int lds0 = (unsigned int)(uintptr_t)(void*)&bsh[0][0];
    const unsigned int lds1 = (unsigned int)(uintptr_t)(void*)&bsh[1][0];

    // prime buffer 0 via TDM (wave 0 only; EXEC-independent DMA, TENSORcnt-tracked)
    if (wave == 0) {
        tdm_load_tile(fb + (size_t)(jt0 * 16) * DD, lds0);
        __builtin_amdgcn_s_wait_tensorcnt(0);
    }
    __syncthreads();

    for (int c = 0; c < CTILES; ++c) {
        const int cur = c & 1;
        if (wave == 0 && (c + 1) < CTILES)
            tdm_load_tile(fb + (size_t)((jt0 + c + 1) * 16) * DD, cur ? lds0 : lds1);

        // B fragment source: column = ln, K = kb + 16g + [0..15] (contiguous, same K order as A)
        const unsigned short* bcol = bsh[cur] + ln * LDS_STRIDE + 16 * g;

        // software-pipelined K loop, 2-deep rotating register buffer:
        // WMMA kk consumes bf[kk&1]; the loads for kk+2 immediately refill it,
        // so pair kk+1 is always in flight behind the current WMMA.
        BFrag bf[2];
        bf[0].h.lo = *(const v8u*)(bcol);
        bf[0].h.hi = *(const v8u*)(bcol + 8);
        bf[1].h.lo = *(const v8u*)(bcol + 32);
        bf[1].h.hi = *(const v8u*)(bcol + 40);
        v8f acc_c = {0.f, 0.f, 0.f, 0.f, 0.f, 0.f, 0.f, 0.f};
        #pragma unroll
        for (int kk = 0; kk < KSTEPS; ++kk) {
            acc_c = __builtin_amdgcn_wmma_f32_16x16x32_bf16(
                        false, afrag[kk], false, bf[kk & 1].b, (short)0, acc_c,
                        false, false);
            if (kk + 2 < KSTEPS) {
                const unsigned short* bp = bcol + (kk + 2) * 32;
                bf[kk & 1].h.lo = *(const v8u*)(bp);
                bf[kk & 1].h.hi = *(const v8u*)(bp + 8);
            }
        }
        // Pin the pipelined shape: 4 DS-reads (prologue), then 14 x (WMMA + 2
        // DS-reads), then the final 2 WMMAs.  0x008 = MFMA/WMMA, 0x100 = DS read.
        __builtin_amdgcn_sched_group_barrier(0x100, 4, 0);
        #pragma unroll
        for (int kk = 0; kk < KSTEPS - 2; ++kk) {
            __builtin_amdgcn_sched_group_barrier(0x008, 1, 0);
            __builtin_amdgcn_sched_group_barrier(0x100, 2, 0);
        }
        __builtin_amdgcn_sched_group_barrier(0x008, 2, 0);

        // streaming epilogue: |logit| <= 10 => plain exp-sum, no max tracking
        const int j = (jt0 + c) * 16 + ln;
        #pragma unroll
        for (int r = 0; r < 8; ++r) {
            const int i = I0 + r + 8 * g;        // C layout: M = r + 8g
            const float s = acc_c[r];
            const float e = __expf(s * INVT);
            acc[r] += e;
            const int p = (i < BB) ? (i + BB) : (i - BB);
            if (j == p) pos[r] = s;              // raw partner similarity
            if (j == i) dia[r] = e;              // diagonal exp term
        }

        if (wave == 0 && (c + 1) < CTILES)
            __builtin_amdgcn_s_wait_tensorcnt(0);
        __syncthreads();
    }

    // 16-lane butterfly reduce; wave exclusively owns its (split, 16-rows) slice
    float* rowsum = rowsum_p + (size_t)split * NROW;
    float* posv   = posv_p   + (size_t)split * NROW;
    float* diagv  = diagv_p  + (size_t)split * NROW;
    #pragma unroll
    for (int r = 0; r < 8; ++r) {
        float a = acc[r], pv = pos[r], dv = dia[r];
        #pragma unroll
        for (int m = 8; m >= 1; m >>= 1) {
            a  += __shfl_xor(a,  m, 32);
            pv += __shfl_xor(pv, m, 32);
            dv += __shfl_xor(dv, m, 32);
        }
        if (ln == 0) {
            const int i = I0 + r + 8 * g;
            rowsum[i] = a;
            posv[i]   = pv;
            diagv[i]  = dv;
        }
    }
}

// ---------------- Kernel 3: fold splits, per-row loss, mean ----------------
__global__ void k_loss(const float* __restrict__ rowsum_p,
                       const float* __restrict__ posv_p,
                       const float* __restrict__ diagv_p,
                       const float* __restrict__ w,
                       float* __restrict__ out) {
    __shared__ float red[256];
    const int t = threadIdx.x;
    float local = 0.f;
    for (int i = t; i < NROW; i += 256) {
        float rs = 0.f, sp = 0.f, dv = 0.f;
        #pragma unroll
        for (int s = 0; s < SPLIT; ++s) {
            rs += rowsum_p[(size_t)s * NROW + i];
            sp += posv_p[(size_t)s * NROW + i];   // exactly one split is nonzero
            dv += diagv_p[(size_t)s * NROW + i];
        }
        const float posi = (w[i] >= 0.f) ? sp : -sp;  // pos = sign(w) * s(i,partner)
        // negatives = all - diag - partner; then add the positive logit
        const float total = rs - dv - __expf(sp * INVT) + __expf(posi * INVT);
        local += __logf(total) - posi * INVT;         // logsumexp - pos/T
    }
    red[t] = local;
    __syncthreads();
    #pragma unroll
    for (int s = 128; s > 0; s >>= 1) {
        if (t < s) red[t] += red[t + s];
        __syncthreads();
    }
    if (t == 0) out[0] = red[0] / (float)NROW;
}

extern "C" void kernel_launch(void* const* d_in, const int* in_sizes, int n_in,
                              void* d_out, int out_size, void* d_ws, size_t ws_size,
                              hipStream_t stream) {
    const float* features = (const float*)d_in[0];   // [8192, 512] f32
    const float* w        = (const float*)d_in[1];   // [8192, 1]   f32

    unsigned short* fb = (unsigned short*)d_ws;                    // 8 MB bf16
    float* rowsum_p = (float*)((char*)d_ws + (size_t)NROW * DD * 2);
    float* posv_p   = rowsum_p + (size_t)SPLIT * NROW;
    float* diagv_p  = posv_p   + (size_t)SPLIT * NROW;

    k_normalize<<<NROW, 256, 0, stream>>>(features, fb);
    k_gram_lse<<<64 * SPLIT, 256, 0, stream>>>(fb, rowsum_p, posv_p, diagv_p);
    k_loss<<<1, 256, 0, stream>>>(rowsum_p, posv_p, diagv_p, w, (float*)d_out);
}
